// AttentionalAggregation_30786325578422
// MI455X (gfx1250) — compile-verified, matrix operand order checked by reading, not run on hardware
//
#include <hip/hip_runtime.h>

// ---------------- problem constants (match reference) ----------------
#define IN_DIM 128
#define OUT_DIM 512
#define LANES_PER_OBS 16
#define KP (IN_DIM + 8)      // padded LDS row stride (bf16 elems); 272B rows -> <=2-way conflicts
#define WAVES 16             // 512 threads per block
#define COLS_PER_BLOCK (WAVES * 16)   // 256 columns -> grid.y = 2
#define OBS_PER_BLOCK 16     // obstacles streamed per block

typedef __attribute__((ext_vector_type(16))) __bf16          v16bf;
typedef __attribute__((ext_vector_type(8)))  float           v8f;
typedef __attribute__((ext_vector_type(8)))  unsigned short  u16x8;
typedef __attribute__((ext_vector_type(16))) unsigned short  u16x16;
typedef __attribute__((ext_vector_type(4)))  unsigned short  u16x4;

// ---- bf16 split helpers: x = hi + lo (both RNE bf16), A*B ~ AhiBhi + AhiBlo + AloBhi ----
__device__ __forceinline__ unsigned short f32_to_bf16_rne(float x) {
    unsigned int u = __float_as_uint(x);
    unsigned int r = u + 0x7FFFu + ((u >> 16) & 1u);
    return (unsigned short)(r >> 16);
}
__device__ __forceinline__ float bf16_to_f32(unsigned short h) {
    return __uint_as_float(((unsigned int)h) << 16);
}

__device__ __forceinline__ v16bf make_frag(u16x8 lo8, u16x8 hi8) {
    u16x16 t;
#pragma unroll
    for (int i = 0; i < 8; ++i) { t[i] = lo8[i]; t[8 + i] = hi8[i]; }
    return __builtin_bit_cast(v16bf, t);
}

// ---------------- prep: W [512,128] f32 -> bf16 hi/lo planes in workspace ----------------
__global__ __launch_bounds__(256) void prep_w_kernel(
    const float* __restrict__ W, unsigned short* __restrict__ whi,
    unsigned short* __restrict__ wlo)
{
    const int idx = (blockIdx.x * 256 + threadIdx.x) * 4;   // 65536 floats total
    float4 v = *(const float4*)(W + idx);
    float f[4] = {v.x, v.y, v.z, v.w};
    u16x4 hi, lo;
#pragma unroll
    for (int j = 0; j < 4; ++j) {
        unsigned short hb = f32_to_bf16_rne(f[j]);
        hi[j] = hb;
        lo[j] = f32_to_bf16_rne(f[j] - bf16_to_f32(hb));
    }
    *(u16x4*)(whi + idx) = hi;
    *(u16x4*)(wlo + idx) = lo;
}

// ---- stage one obstacle's A tile (16x128 f32) into bf16 hi/lo LDS planes ----
__device__ __forceinline__ void stage_A(const float* __restrict__ src_base,
                                        unsigned short* __restrict__ dAhi,
                                        unsigned short* __restrict__ dAlo,
                                        int tid)
{
    const int idx = tid * 4;             // 2048 floats / 512 threads = one float4 each
    const int row = idx >> 7;
    const int k   = idx & 127;
    float4 v = *(const float4*)(src_base + idx);
    // prefetch the obstacle-after-next's corresponding cacheline
    __builtin_prefetch(src_base + 2 * LANES_PER_OBS * IN_DIM + idx, 0, 0);
    float f[4] = {v.x, v.y, v.z, v.w};
    u16x4 hi, lo;
#pragma unroll
    for (int j = 0; j < 4; ++j) {
        unsigned short hb = f32_to_bf16_rne(f[j]);
        hi[j] = hb;
        lo[j] = f32_to_bf16_rne(f[j] - bf16_to_f32(hb));
    }
    *(u16x4*)&dAhi[row * KP + k] = hi;
    *(u16x4*)&dAlo[row * KP + k] = lo;
}

// ---------------- main fused kernel ----------------
// B (W hi/lo fragments for this wave's 16 columns) lives in registers for the whole block.
// USE_WS=1: fragments loaded from precomputed ws planes; USE_WS=0: converted from f32 W.
template <bool USE_WS>
__global__ __launch_bounds__(512) void fused_lane_gemm_pool(
    const float* __restrict__ lane_enc,           // [M, 128]
    const float* __restrict__ W,                  // [512, 128] row-major (out, in)
    const unsigned short* __restrict__ whi,       // [512*128] bf16 hi plane (ws)
    const unsigned short* __restrict__ wlo,       // [512*128] bf16 lo plane (ws)
    const float* __restrict__ bias,               // [512]
    float* __restrict__ out,                      // [N_OBS, 1024] = [max(512) | avg(512)]
    int n_obs)
{
    __shared__ unsigned short sAhi[2][LANES_PER_OBS * KP];
    __shared__ unsigned short sAlo[2][LANES_PER_OBS * KP];

    const int tid  = threadIdx.x;                 // 0..511 (16 wave32s)
    const int wave = tid >> 5;
    const int lane = tid & 31;
    const int m    = lane & 15;                   // A row / B column within 16x16 tile
    const int half = lane >> 4;

    const int n0 = blockIdx.y * COLS_PER_BLOCK + wave * 16;   // this wave's column base
    const int n  = n0 + m;                                    // lane's column == W row

    // ---------- load this wave's B fragments into registers (once per block) ----------
    v16bf Bhi[4], Blo[4];
    if (USE_WS) {
        const unsigned short* hs = whi + (size_t)n * IN_DIM;
        const unsigned short* ls = wlo + (size_t)n * IN_DIM;
#pragma unroll
        for (int c = 0; c < 4; ++c) {
            const int kb = c * 32 + half * 8;
            Bhi[c] = make_frag(*(const u16x8*)(hs + kb), *(const u16x8*)(hs + kb + 16));
            Blo[c] = make_frag(*(const u16x8*)(ls + kb), *(const u16x8*)(ls + kb + 16));
        }
    } else {
        const float* wsrc = W + (size_t)n * IN_DIM;
#pragma unroll
        for (int c = 0; c < 4; ++c) {
            const int kb = c * 32 + half * 8;
            u16x8 h0, h1, l0, l1;
#pragma unroll
            for (int j = 0; j < 8; ++j) {
                float x0 = wsrc[kb + j];
                float x1 = wsrc[kb + 16 + j];
                unsigned short hb0 = f32_to_bf16_rne(x0);
                unsigned short hb1 = f32_to_bf16_rne(x1);
                h0[j] = hb0; l0[j] = f32_to_bf16_rne(x0 - bf16_to_f32(hb0));
                h1[j] = hb1; l1[j] = f32_to_bf16_rne(x1 - bf16_to_f32(hb1));
            }
            Bhi[c] = make_frag(h0, h1);
            Blo[c] = make_frag(l0, l1);
        }
    }
    const float bn = bias[n];

    // ---------- stream a strip of obstacles through double-buffered A tiles ----------
    const int obs0  = blockIdx.x * OBS_PER_BLOCK;
    const int count = min(OBS_PER_BLOCK, n_obs - obs0);
    if (count <= 0) return;

    const float* Asrc = lane_enc + (size_t)obs0 * (LANES_PER_OBS * IN_DIM);

    stage_A(Asrc, sAhi[0], sAlo[0], tid);

    for (int i = 0; i < count; ++i) {
        __syncthreads();   // buf[i&1] ready; previous compute (same parity writer) done
        if (i + 1 < count)
            stage_A(Asrc + (size_t)(i + 1) * (LANES_PER_OBS * IN_DIM),
                    sAhi[(i + 1) & 1], sAlo[(i + 1) & 1], tid);

        const unsigned short* ah = &sAhi[i & 1][m * KP];
        const unsigned short* al = &sAlo[i & 1][m * KP];

        v8f acc = {};
#pragma unroll
        for (int c = 0; c < 4; ++c) {
            const int kb = c * 32 + half * 8;
            v16bf Ahi = make_frag(*(const u16x8*)(ah + kb), *(const u16x8*)(ah + kb + 16));
            v16bf Alo = make_frag(*(const u16x8*)(al + kb), *(const u16x8*)(al + kb + 16));

            acc = __builtin_amdgcn_wmma_f32_16x16x32_bf16(false, Ahi, false, Bhi[c],
                                                          (short)0, acc, false, false);
            acc = __builtin_amdgcn_wmma_f32_16x16x32_bf16(false, Ahi, false, Blo[c],
                                                          (short)0, acc, false, false);
            acc = __builtin_amdgcn_wmma_f32_16x16x32_bf16(false, Alo, false, Bhi[c],
                                                          (short)0, acc, false, false);
        }

        // ---- epilogue: bias + ReLU + group max/mean (group == the 16-row M tile) ----
        // C layout: lanes 0-15: VGPR v -> (M=v, N=lane); lanes 16-31: VGPR v -> (M=v+8, N=lane-16)
        float mx = -__builtin_inff();
        float sm = 0.0f;
#pragma unroll
        for (int v = 0; v < 8; ++v) {
            float e = acc[v] + bn;
            e = fmaxf(e, 0.0f);
            mx = fmaxf(mx, e);
            sm += e;
        }
        mx = fmaxf(mx, __shfl_xor(mx, 16, 32));
        sm += __shfl_xor(sm, 16, 32);

        float* obase = out + (size_t)(obs0 + i) * (2 * OUT_DIM);
        if (lane < 16) {
            obase[n0 + lane] = mx;                                    // max  -> cols [0,512)
        } else {
            obase[OUT_DIM + n0 + (lane - 16)] = sm * (1.0f / 16.0f);  // mean -> cols [512,1024)
        }
    }
}

extern "C" void kernel_launch(void* const* d_in, const int* in_sizes, int n_in,
                              void* d_out, int out_size, void* d_ws, size_t ws_size,
                              hipStream_t stream) {
    // setup_inputs order: obs_encoding, lane_encoding, same_obs_mask, W, b
    const float* lane_enc = (const float*)d_in[1];
    const float* W        = (const float*)d_in[3];
    const float* b        = (const float*)d_in[4];
    float* out            = (float*)d_out;

    const int n_obs = in_sizes[0] / IN_DIM;       // 25000 (group size fixed at 16)

    const size_t plane = (size_t)OUT_DIM * IN_DIM;              // 65536 elems
    const size_t need  = 2 * plane * sizeof(unsigned short);    // 256 KB

    const int nStrips = (n_obs + OBS_PER_BLOCK - 1) / OBS_PER_BLOCK;
    dim3 grid((unsigned)nStrips, OUT_DIM / COLS_PER_BLOCK, 1);  // (1563, 2)

    if (ws_size >= need) {
        unsigned short* whi = (unsigned short*)d_ws;
        unsigned short* wlo = whi + plane;
        prep_w_kernel<<<plane / (256 * 4), 256, 0, stream>>>(W, whi, wlo);
        fused_lane_gemm_pool<true><<<grid, 512, 0, stream>>>(
            lane_enc, W, whi, wlo, b, out, n_obs);
    } else {
        fused_lane_gemm_pool<false><<<grid, 512, 0, stream>>>(
            lane_enc, W, nullptr, nullptr, b, out, n_obs);
    }
    (void)n_in; (void)out_size; (void)in_sizes;
}
